// Propgate_time_step_50586124812595
// MI455X (gfx1250) — compile-verified, matrix operand order checked by reading, not run on hardware
//
#include <hip/hip_runtime.h>
#include <hip/hip_bf16.h>

// ---------------- problem constants ----------------
#define HID   256
#define WHAT  50
#define WHERE 3
#define ENCD  100
#define GDIM  20
#define IMH   50
#define IMW   50
#define NB    4096
#define TNUM  8

#define KREL  736   // 462 (x) + 256 (h) = 718 -> pad 736 (multiple of 32)
#define KTEM  896   // 615 (x) + 256 (h) = 871 -> pad 896
#define KGL   416   // 400 -> pad 416
#define NGL   128   // 100 -> pad 128 (multiple of 64 for 2x2-tiled GEMM)
#define NG    1024  // 4*HID gate width

typedef __attribute__((ext_vector_type(16))) _Float16 v16h;
typedef __attribute__((ext_vector_type(8)))  _Float16 v8h;
typedef __attribute__((ext_vector_type(8)))  float    v8f;

__device__ __forceinline__ float sigf(float x)      { return 1.0f / (1.0f + __expf(-x)); }
__device__ __forceinline__ float softplusf(float x) { return (x > 20.0f) ? x : log1pf(__expf(x)); }

__device__ __forceinline__ v16h ldfrag(const _Float16* p)
{
    v8h lo = *(const v8h*)(p);
    v8h hi = *(const v8h*)(p + 16);
    return __builtin_shufflevector(lo, hi, 0,1,2,3,4,5,6,7,8,9,10,11,12,13,14,15);
}

// ---------------- WMMA TN GEMM, 2x2 register tiling ----------------
// C[M, Nout] = A[M, K] * B[Nout, K]^T  (+bias, optional relu)
// A, B: f16 row-major, K-contiguous (K multiple of 32, 16B aligned).
// Each wave computes a 32x32 C macro-tile (4 WMMA accumulators); a 256-thread
// block = 8 waves arranged 4(M) x 2(N) -> block tile 128(M) x 64(N).
// Requires M % 128 == 0, Nout % 64 == 0.
// A-operand lane layout (ISA 7.12.2, 16-bit A 16x32): lane L holds row L%16,
// K = hs..hs+7 and 16+hs..16+hs+7 with hs = (L/16)*8. B mirrors A (rows of W).
__global__ __launch_bounds__(256)
void wmma_gemm_tn(const _Float16* __restrict__ A, const _Float16* __restrict__ B,
                  const float* __restrict__ bias, float* __restrict__ C,
                  int M, int Nout, int K, int relu)
{
    const int lane = threadIdx.x & 31;
    const int wave = threadIdx.x >> 5;
    const int wm   = wave >> 1;                  // 0..3
    const int wn   = wave & 1;                   // 0..1
    const int l15  = lane & 15;
    const int hs   = (lane >> 4) * 8;            // K half-select offset

    const int m0 = blockIdx.y * 128 + wm * 32;   // wave's M base
    const int n0 = blockIdx.x * 64  + wn * 32;   // wave's N base

    const _Float16* pa0 = A + (size_t)(m0 + l15)      * K + hs;
    const _Float16* pa1 = A + (size_t)(m0 + 16 + l15) * K + hs;
    const _Float16* pb0 = B + (size_t)(n0 + l15)      * K + hs;
    const _Float16* pb1 = B + (size_t)(n0 + 16 + l15) * K + hs;

    v8f acc00 = {}, acc01 = {}, acc10 = {}, acc11 = {};
    for (int k = 0; k < K; k += 32) {
        v16h a0 = ldfrag(pa0 + k);
        v16h a1 = ldfrag(pa1 + k);
        v16h b0 = ldfrag(pb0 + k);
        v16h b1 = ldfrag(pb1 + k);
        acc00 = __builtin_amdgcn_wmma_f32_16x16x32_f16(false, a0, false, b0, (short)0, acc00, false, false);
        acc01 = __builtin_amdgcn_wmma_f32_16x16x32_f16(false, a0, false, b1, (short)0, acc01, false, false);
        acc10 = __builtin_amdgcn_wmma_f32_16x16x32_f16(false, a1, false, b0, (short)0, acc10, false, false);
        acc11 = __builtin_amdgcn_wmma_f32_16x16x32_f16(false, a1, false, b1, (short)0, acc11, false, false);
    }

    // C layout: vgpr v -> M = mtile + v + (lane/16)*8 ; N = ntile + lane%16
    const int mb   = m0 + (lane >> 4) * 8;
    const int col0 = n0 + l15;
    const int col1 = col0 + 16;
    float b0add = bias ? bias[col0] : 0.0f;
    float b1add = bias ? bias[col1] : 0.0f;
#pragma unroll
    for (int v = 0; v < 8; ++v) {
        float x00 = acc00[v] + b0add, x01 = acc01[v] + b1add;
        float x10 = acc10[v] + b0add, x11 = acc11[v] + b1add;
        if (relu) {
            x00 = fmaxf(x00, 0.0f); x01 = fmaxf(x01, 0.0f);
            x10 = fmaxf(x10, 0.0f); x11 = fmaxf(x11, 0.0f);
        }
        C[(size_t)(mb + v)      * Nout + col0] = x00;
        C[(size_t)(mb + v)      * Nout + col1] = x01;
        C[(size_t)(mb + 16 + v) * Nout + col0] = x10;
        C[(size_t)(mb + 16 + v) * Nout + col1] = x11;
    }
}

// ---------------- weight packing (f32 -> padded f16) ----------------
__global__ void pack_wrel(const float* __restrict__ Wih, const float* __restrict__ Whh,
                          _Float16* __restrict__ W)
{
    int idx = blockIdx.x * blockDim.x + threadIdx.x;
    if (idx >= NG * KREL) return;
    int r = idx / KREL, c = idx % KREL;
    float v = (c < 462) ? Wih[(size_t)r * 462 + c]
            : (c < 718) ? Whh[(size_t)r * 256 + (c - 462)] : 0.0f;
    W[idx] = (_Float16)v;
}

__global__ void pack_wtem(const float* __restrict__ Wih, const float* __restrict__ Whh,
                          _Float16* __restrict__ W)
{
    int idx = blockIdx.x * blockDim.x + threadIdx.x;
    if (idx >= NG * KTEM) return;
    int r = idx / KTEM, c = idx % KTEM;
    float v = (c < 615) ? Wih[(size_t)r * 615 + c]
            : (c < 871) ? Whh[(size_t)r * 256 + (c - 615)] : 0.0f;
    W[idx] = (_Float16)v;
}

__global__ void pack_wgl(const float* __restrict__ glW, const float* __restrict__ glb,
                         _Float16* __restrict__ W, float* __restrict__ bpad)
{
    int idx = blockIdx.x * blockDim.x + threadIdx.x;
    if (idx >= NGL * KGL) return;
    int r = idx / KGL, c = idx % KGL;
    W[idx] = (_Float16)((r < ENCD && c < 400) ? glW[(size_t)r * 400 + c] : 0.0f);
    if (c == 0) bpad[r] = (r < ENCD) ? glb[r] : 0.0f;
}

// ---------------- loc head: z_where_bias = relu(hid @ locW^T + b) + where_l ----
__global__ void loc_kernel(const float* __restrict__ hid_last,
                           const float* __restrict__ where_last,
                           const float* __restrict__ locW, const float* __restrict__ locb,
                           float* __restrict__ zwb, int t)
{
    int n = blockIdx.x * blockDim.x + threadIdx.x;
    if (n >= NB) return;
    const float* h  = hid_last   + ((size_t)n * TNUM + t) * HID;
    const float* wl = where_last + ((size_t)n * TNUM + t) * WHERE;
    for (int j = 0; j < WHERE; ++j) {
        const float* w = locW + (size_t)j * HID;
        float acc = locb[j];
        for (int k = 0; k < HID; ++k) acc += h[k] * w[k];
        zwb[n * WHERE + j] = fmaxf(acc, 0.0f) + wl[j];
    }
}

// ---------------- STN bilinear sampler -> f16 padded patch [NB, KGL] ----------
__global__ void stn_kernel(const float* __restrict__ img, const float* __restrict__ zw,
                           _Float16* __restrict__ patch)
{
    int idx = blockIdx.x * blockDim.x + threadIdx.x;
    if (idx >= NB * KGL) return;
    int n = idx / KGL, p = idx % KGL;
    if (p >= GDIM * GDIM) { patch[idx] = (_Float16)0.0f; return; }
    int py = p / GDIM, px = p % GDIM;
    float gx = ((px + 0.5f) / (float)GDIM) * 2.0f - 1.0f;
    float gy = ((py + 0.5f) / (float)GDIM) * 2.0f - 1.0f;
    float s  = zw[n * 3 + 0], tx = zw[n * 3 + 1], ty = zw[n * 3 + 2];
    float ix = ((s * gx + tx + 1.0f) * (float)IMW - 1.0f) * 0.5f;
    float iy = ((s * gy + ty + 1.0f) * (float)IMH - 1.0f) * 0.5f;
    float x0f = floorf(ix), y0f = floorf(iy);
    float wx1 = ix - x0f,   wy1 = iy - y0f;
    int x0 = (int)x0f, y0 = (int)y0f;
    const float* im = img + (size_t)n * IMH * IMW;
    auto gat = [&](int yy, int xx) -> float {
        bool valid = (xx >= 0) & (xx < IMW) & (yy >= 0) & (yy < IMH);
        int yc = yy < 0 ? 0 : (yy > IMH - 1 ? IMH - 1 : yy);
        int xc = xx < 0 ? 0 : (xx > IMW - 1 ? IMW - 1 : xx);
        float v = im[yc * IMW + xc];
        return valid ? v : 0.0f;
    };
    float out = (1.0f - wy1) * ((1.0f - wx1) * gat(y0, x0)     + wx1 * gat(y0, x0 + 1))
              +         wy1  * ((1.0f - wx1) * gat(y0 + 1, x0) + wx1 * gat(y0 + 1, x0 + 1));
    patch[idx] = (_Float16)out;
}

// ---------------- concat packers (f32 sources -> f16 padded X) ----------------
__global__ void pack_rel(const float* __restrict__ enc,  const float* __restrict__ where_last,
                         const float* __restrict__ zw_prev, const float* __restrict__ what_last,
                         const float* __restrict__ zwhat_prev, const float* __restrict__ hid_last,
                         const float* __restrict__ h_rela, _Float16* __restrict__ X, int t)
{
    int idx = blockIdx.x * blockDim.x + threadIdx.x;
    if (idx >= NB * KREL) return;
    int n = idx / KREL, c = idx % KREL;
    float v;
    if      (c < 100) v = enc[(size_t)n * NGL + c];
    else if (c < 103) v = where_last[((size_t)n * TNUM + t) * WHERE + (c - 100)];
    else if (c < 106) v = zw_prev[n * WHERE + (c - 103)];
    else if (c < 156) v = what_last[((size_t)n * TNUM + t) * WHAT + (c - 106)];
    else if (c < 206) v = zwhat_prev[n * WHAT + (c - 156)];
    else if (c < 462) v = hid_last[((size_t)n * TNUM + t) * HID + (c - 206)];
    else if (c < 718) v = h_rela[(size_t)n * HID + (c - 462)];
    else              v = 0.0f;
    X[idx] = (_Float16)v;
}

__global__ void pack_tem(const float* __restrict__ enc, const float* __restrict__ zw_prev,
                         const float* __restrict__ hid_last, const float* __restrict__ h_rela,
                         _Float16* __restrict__ X, int t)
{
    int idx = blockIdx.x * blockDim.x + threadIdx.x;
    if (idx >= NB * KTEM) return;
    int n = idx / KTEM, c = idx % KTEM;
    float v;
    if      (c < 100) v = enc[(size_t)n * NGL + c];
    else if (c < 103) v = zw_prev[n * WHERE + (c - 100)];
    else if (c < 359) v = hid_last[((size_t)n * TNUM + t) * HID + (c - 103)];
    else if (c < 615) v = h_rela[(size_t)n * HID + (c - 359)];
    else if (c < 871) v = h_rela[(size_t)n * HID + (c - 615)];  // recurrent h for tem Whh
    else              v = 0.0f;
    X[idx] = (_Float16)v;
}

// ---------------- LSTM pointwise ----------------
__global__ void rel_pointwise(const float* __restrict__ Gm, const float* __restrict__ b,
                              float* __restrict__ h, float* __restrict__ c)
{
    int idx = blockIdx.x * blockDim.x + threadIdx.x;
    if (idx >= NB * HID) return;
    int n = idx / HID, j = idx % HID;
    const float* g = Gm + (size_t)n * NG;
    float gi = g[j]       + b[j];
    float gf = g[j + 256] + b[j + 256];
    float gg = g[j + 512] + b[j + 512];
    float go = g[j + 768] + b[j + 768];
    float c2 = sigf(gf) * c[idx] + sigf(gi) * tanhf(gg);
    c[idx] = c2;
    h[idx] = sigf(go) * tanhf(c2);
}

__global__ void tem_pointwise(const float* __restrict__ Gm, const float* __restrict__ b,
                              const float* __restrict__ c, float* __restrict__ h_temp,
                              float* __restrict__ out_ht)
{
    int idx = blockIdx.x * blockDim.x + threadIdx.x;
    if (idx >= NB * HID) return;
    int n = idx / HID, j = idx % HID;
    const float* g = Gm + (size_t)n * NG;
    float gi = g[j]       + b[j];
    float gf = g[j + 256] + b[j + 256];
    float gg = g[j + 512] + b[j + 512];
    float go = g[j + 768] + b[j + 768];
    float c2 = sigf(gf) * c[idx] + sigf(gi) * tanhf(gg);
    float hv = sigf(go) * tanhf(c2);
    h_temp[idx] = hv;
    out_ht[idx] = hv;   // overwritten each step; final step leaves htemp[-1]
}

// ---------------- whr head: z_where sample ----------------
__global__ void whr_kernel(const float* __restrict__ where_last, const float* __restrict__ h_rela,
                           const float* __restrict__ W, const float* __restrict__ bb,
                           const float* __restrict__ eps_where, float* __restrict__ zw_cur,
                           float* __restrict__ out_zwhere, int t)
{
    int n = blockIdx.x * blockDim.x + threadIdx.x;
    if (n >= NB) return;
    const float* wl = where_last + ((size_t)n * TNUM + t) * WHERE;
    const float* h  = h_rela + (size_t)n * HID;
    float ms[6];
    for (int j = 0; j < 6; ++j) {
        const float* w = W + (size_t)j * 259;
        float acc = bb[j] + wl[0] * w[0] + wl[1] * w[1] + wl[2] * w[2];
        for (int k = 0; k < HID; ++k) acc += h[k] * w[3 + k];
        ms[j] = acc;
    }
    const float* ew = eps_where + ((size_t)t * NB + n) * WHERE;
    for (int j = 0; j < WHERE; ++j) {
        float z = ms[j] + ew[j] * softplusf(ms[3 + j]);
        zw_cur[n * WHERE + j] = z;
        out_zwhere[((size_t)n * TNUM + t) * WHERE + j] = z;
    }
}

// ---------------- wht head: z_what sample ----------------
__global__ void wht_kernel(const float* __restrict__ what_last, const float* __restrict__ h_rela,
                           const float* __restrict__ h_temp, const float* __restrict__ W,
                           const float* __restrict__ bb, const float* __restrict__ eps_what,
                           float* __restrict__ zwhat_cur, float* __restrict__ out_zwhat, int t)
{
    int idx = blockIdx.x * blockDim.x + threadIdx.x;
    if (idx >= NB * WHAT) return;
    int n = idx / WHAT, j = idx % WHAT;
    const float* wl = what_last + ((size_t)n * TNUM + t) * WHAT;
    const float* hr = h_rela + (size_t)n * HID;
    const float* ht = h_temp + (size_t)n * HID;
    const float* wm = W + (size_t)j * 562;
    const float* ws = W + (size_t)(j + WHAT) * 562;
    float m = bb[j], s = bb[j + WHAT];
    for (int k = 0; k < WHAT; ++k) { m += wl[k] * wm[k];        s += wl[k] * ws[k]; }
    for (int k = 0; k < HID;  ++k) { m += hr[k] * wm[50 + k];   s += hr[k] * ws[50 + k]; }
    for (int k = 0; k < HID;  ++k) { m += ht[k] * wm[306 + k];  s += ht[k] * ws[306 + k]; }
    float z = m + eps_what[((size_t)t * NB + n) * WHAT + j] * softplusf(s);
    zwhat_cur[n * WHAT + j] = z;
    out_zwhat[((size_t)n * TNUM + t) * WHAT + j] = z;
}

// ---------------- pres head ----------------
__global__ void pres_kernel(const float* __restrict__ zwhat_cur, const float* __restrict__ zw_cur,
                            const float* __restrict__ h_rela, const float* __restrict__ h_temp,
                            const float* __restrict__ prsW, const float* __restrict__ prsb,
                            const float* __restrict__ u_pres, const float* __restrict__ pres_last,
                            float* __restrict__ out_zpres, int t)
{
    int n = blockIdx.x * blockDim.x + threadIdx.x;
    if (n >= NB) return;
    float acc = prsb[0];
    for (int k = 0; k < WHAT;  ++k) acc += zwhat_cur[n * WHAT + k]  * prsW[k];
    for (int k = 0; k < WHERE; ++k) acc += zw_cur[n * WHERE + k]    * prsW[50 + k];
    for (int k = 0; k < HID;   ++k) acc += h_rela[(size_t)n * HID + k] * prsW[53 + k];
    for (int k = 0; k < HID;   ++k) acc += h_temp[(size_t)n * HID + k] * prsW[309 + k];
    float para = sigf(acc);
    float u  = u_pres[(size_t)t * NB + n];
    float pl = pres_last[(size_t)n * TNUM + t];
    out_zpres[(size_t)n * TNUM + t] = (u < para ? 1.0f : 0.0f) * pl;
}

// ---------------- host ----------------
static inline size_t alignup(size_t x) { return (x + 255) & ~(size_t)255; }

extern "C" void kernel_launch(void* const* d_in, const int* in_sizes, int n_in,
                              void* d_out, int out_size, void* d_ws, size_t ws_size,
                              hipStream_t stream)
{
    // inputs (setup_inputs dict order, params dict flattened in insertion order)
    const float* img        = (const float*)d_in[0];
    const float* what_last  = (const float*)d_in[1];
    const float* where_last = (const float*)d_in[2];
    const float* pres_last  = (const float*)d_in[3];
    const float* hid_last   = (const float*)d_in[4];
    const float* eps_where  = (const float*)d_in[5];
    const float* eps_what   = (const float*)d_in[6];
    const float* u_pres     = (const float*)d_in[7];
    const float* rel_Wih = (const float*)d_in[8];
    const float* rel_Whh = (const float*)d_in[9];
    const float* rel_b   = (const float*)d_in[10];
    const float* tem_Wih = (const float*)d_in[11];
    const float* tem_Whh = (const float*)d_in[12];
    const float* tem_b   = (const float*)d_in[13];
    const float* loc_W   = (const float*)d_in[14];
    const float* loc_b   = (const float*)d_in[15];
    const float* gl_W    = (const float*)d_in[16];
    const float* gl_b    = (const float*)d_in[17];
    const float* whr_W   = (const float*)d_in[18];
    const float* whr_b   = (const float*)d_in[19];
    const float* wht_W   = (const float*)d_in[20];
    const float* wht_b   = (const float*)d_in[21];
    const float* prs_W   = (const float*)d_in[22];
    const float* prs_b   = (const float*)d_in[23];

    // workspace carve-up (~50 MB total)
    char* base = (char*)d_ws;
    size_t off = 0;
    auto take = [&](size_t bytes) { char* p = base + off; off += alignup(bytes); return p; };
    _Float16* Wrel16 = (_Float16*)take((size_t)NG * KREL * 2);
    _Float16* Wtem16 = (_Float16*)take((size_t)NG * KTEM * 2);
    _Float16* Wgl16  = (_Float16*)take((size_t)NGL * KGL * 2);
    float*    glbpad = (float*)   take((size_t)NGL * 4);
    _Float16* Xrel   = (_Float16*)take((size_t)NB * KREL * 2);
    _Float16* Xtem   = (_Float16*)take((size_t)NB * KTEM * 2);
    _Float16* patch  = (_Float16*)take((size_t)NB * KGL * 2);
    float*    enc    = (float*)   take((size_t)NB * NGL * 4);
    float*    Gbuf   = (float*)   take((size_t)NB * NG * 4);
    float*    h_rela = (float*)   take((size_t)NB * HID * 4);
    float*    c_rela = (float*)   take((size_t)NB * HID * 4);
    float*    h_temp = (float*)   take((size_t)NB * HID * 4);
    float*    zwb    = (float*)   take((size_t)NB * WHERE * 4);   // z_where_bias
    float*    zwbuf0 = (float*)   take((size_t)NB * WHERE * 4);
    float*    zwbuf1 = (float*)   take((size_t)NB * WHERE * 4);
    float*    zwtbuf0= (float*)   take((size_t)NB * WHAT * 4);
    float*    zwtbuf1= (float*)   take((size_t)NB * WHAT * 4);
    (void)ws_size; (void)in_sizes; (void)n_in; (void)out_size;

    // output regions (reference return order, flattened)
    float* out_zwhat  = (float*)d_out;                          // [NB, TNUM, WHAT]
    float* out_zwhere = out_zwhat  + (size_t)NB * TNUM * WHAT;  // [NB, TNUM, WHERE]
    float* out_zpres  = out_zwhere + (size_t)NB * TNUM * WHERE; // [NB, TNUM, 1]
    float* out_htemp  = out_zpres  + (size_t)NB * TNUM;         // [NB, HID]

    const int BT = 256;
    auto nblk = [&](size_t n) { return (unsigned)((n + BT - 1) / BT); };

    // one-time weight packing (f32 -> padded f16)
    pack_wrel<<<nblk((size_t)NG * KREL), BT, 0, stream>>>(rel_Wih, rel_Whh, Wrel16);
    pack_wtem<<<nblk((size_t)NG * KTEM), BT, 0, stream>>>(tem_Wih, tem_Whh, Wtem16);
    pack_wgl <<<nblk((size_t)NGL * KGL), BT, 0, stream>>>(gl_W, gl_b, Wgl16, glbpad);

    // zero carries (graph-capture-safe async memset)
    hipMemsetAsync(zwbuf0,  0, (size_t)NB * WHERE * 4, stream);
    hipMemsetAsync(zwtbuf0, 0, (size_t)NB * WHAT * 4, stream);
    hipMemsetAsync(h_rela,  0, (size_t)NB * HID * 4, stream);
    hipMemsetAsync(c_rela,  0, (size_t)NB * HID * 4, stream);

    // GEMM grids: block tile = 128(M) x 64(N)
    dim3 gGates(NG / 64, NB / 128), gGl(NGL / 64, NB / 128), blk(BT);

    for (int t = 0; t < TNUM; ++t) {
        float* zw_prev    = (t & 1) ? zwbuf1  : zwbuf0;
        float* zw_cur     = (t & 1) ? zwbuf0  : zwbuf1;
        float* zwhat_prev = (t & 1) ? zwtbuf1 : zwtbuf0;
        float* zwhat_cur  = (t & 1) ? zwtbuf0 : zwtbuf1;

        // z_where_bias -> glimpse(enc_bias)
        loc_kernel<<<nblk(NB), blk, 0, stream>>>(hid_last, where_last, loc_W, loc_b, zwb, t);
        stn_kernel<<<nblk((size_t)NB * KGL), blk, 0, stream>>>(img, zwb, patch);
        wmma_gemm_tn<<<gGl, blk, 0, stream>>>(patch, Wgl16, glbpad, enc, NB, NGL, KGL, 1);

        // rel LSTM (x||h fused GEMM, WMMA) -> h_rela, c_rela
        pack_rel<<<nblk((size_t)NB * KREL), blk, 0, stream>>>(enc, where_last, zw_prev,
                                                              what_last, zwhat_prev, hid_last,
                                                              h_rela, Xrel, t);
        wmma_gemm_tn<<<gGates, blk, 0, stream>>>(Xrel, Wrel16, nullptr, Gbuf, NB, NG, KREL, 0);
        rel_pointwise<<<nblk((size_t)NB * HID), blk, 0, stream>>>(Gbuf, rel_b, h_rela, c_rela);

        // z_where sample -> glimpse(enc)
        whr_kernel<<<nblk(NB), blk, 0, stream>>>(where_last, h_rela, whr_W, whr_b,
                                                 eps_where, zw_cur, out_zwhere, t);
        stn_kernel<<<nblk((size_t)NB * KGL), blk, 0, stream>>>(img, zw_cur, patch);
        wmma_gemm_tn<<<gGl, blk, 0, stream>>>(patch, Wgl16, glbpad, enc, NB, NGL, KGL, 1);

        // tem LSTM -> h_temp
        pack_tem<<<nblk((size_t)NB * KTEM), blk, 0, stream>>>(enc, zw_prev, hid_last,
                                                              h_rela, Xtem, t);
        wmma_gemm_tn<<<gGates, blk, 0, stream>>>(Xtem, Wtem16, nullptr, Gbuf, NB, NG, KTEM, 0);
        tem_pointwise<<<nblk((size_t)NB * HID), blk, 0, stream>>>(Gbuf, tem_b, c_rela,
                                                                  h_temp, out_htemp);

        // z_what sample + presence
        wht_kernel<<<nblk((size_t)NB * WHAT), blk, 0, stream>>>(what_last, h_rela, h_temp,
                                                                wht_W, wht_b, eps_what,
                                                                zwhat_cur, out_zwhat, t);
        pres_kernel<<<nblk(NB), blk, 0, stream>>>(zwhat_cur, zw_cur, h_rela, h_temp,
                                                  prs_W, prs_b, u_pres, pres_last,
                                                  out_zpres, t);
    }
}
